// TriMul_47562467836630
// MI455X (gfx1250) — compile-verified, matrix operand order checked by reading, not run on hardware
//
#include <hip/hip_runtime.h>

// ---------------------------------------------------------------------------
// TriMul for MI455X (gfx1250, wave32, WMMA 16x16x32 bf16)
// B=1, N=384, D=H=128.
// ---------------------------------------------------------------------------

#define NN   384
#define DD   128
#define HH   128
#define ROWS (NN * NN)          // 147456

typedef __attribute__((ext_vector_type(16))) __bf16 v16bf;
typedef __attribute__((ext_vector_type(8)))  float  v8f;

// ISA 7.12.2: 16-bit A-matrix 16x32. Per lane (m = lane&15, khalf = lane>>4):
//   VGPR v<4  holds K = v*2   + khalf*8  (and +1)
//   VGPR v>=4 holds K = 16+(v-4)*2 + khalf*8 (and +1)
__device__ __forceinline__ int kbase(int v, int khalf) {
    return (v < 4) ? (v * 2 + khalf * 8) : (16 + (v - 4) * 2 + khalf * 8);
}

__device__ __forceinline__ float sigmoidf_(float x) {
    return 1.0f / (1.0f + __expf(-x));
}

__device__ __forceinline__ v8f wmma_bf16(v16bf a, v16bf b, v8f c) {
    return __builtin_amdgcn_wmma_f32_16x16x32_bf16(
        false, a, false, b, (short)0, c, false, false);
}

// ---------------------------------------------------------------------------
// K1: LayerNorm + 5 projections + mask/sigmoid gating.
//   Per block: one 16-row tile of the flattened [147456, 128] input.
//   8 waves; wave w computes h-columns [16w, 16w+16) for all 5 matrices.
// ---------------------------------------------------------------------------
__global__ __launch_bounds__(256) void trimul_proj_kernel(
    const float* __restrict__ x, const float* __restrict__ mask,
    const float* __restrict__ nw, const float* __restrict__ nb,
    const float* __restrict__ Wl, const float* __restrict__ Wr,
    const float* __restrict__ Wlg, const float* __restrict__ Wrg,
    const float* __restrict__ Wog,
    void* __restrict__ leftp_, void* __restrict__ rightp_,
    float* __restrict__ ogp)
{
    __bf16* __restrict__ leftp  = (__bf16*)leftp_;
    __bf16* __restrict__ rightp = (__bf16*)rightp_;

    __shared__ float  xs[16 * DD];
    __shared__ float  part1[256], part2[256];
    __shared__ float  mu[16], rstd[16];
    __shared__ __bf16 xbf[16 * DD];

    const int  t    = threadIdx.x;
    const int  tile = blockIdx.x;
    const long base = (long)tile * 16 * DD;

    // --- stage 16x128 f32 rows into LDS (coalesced float4) ---
    {
        const float4* g = (const float4*)(x + base);
        float4*       s = (float4*)xs;
        s[t]       = g[t];
        s[t + 256] = g[t + 256];
    }
    __syncthreads();

    // --- LayerNorm: 16 lanes per row, 8 elements each ---
    {
        const int row = t >> 4, seg = t & 15;
        float s = 0.f, s2 = 0.f;
#pragma unroll
        for (int e = 0; e < 8; ++e) {
            float v = xs[row * DD + seg * 8 + e];
            s += v; s2 += v * v;
        }
        part1[row * 16 + seg] = s;
        part2[row * 16 + seg] = s2;
    }
    __syncthreads();
    if (t < 16) {
        float s = 0.f, s2 = 0.f;
#pragma unroll
        for (int e = 0; e < 16; ++e) { s += part1[t * 16 + e]; s2 += part2[t * 16 + e]; }
        float m   = s  * (1.0f / 128.0f);
        float var = s2 * (1.0f / 128.0f) - m * m;
        mu[t]   = m;
        rstd[t] = rsqrtf(var + 1e-5f);
    }
    __syncthreads();
    {
        const int row = t >> 4, seg = t & 15;
        const float m = mu[row], rs = rstd[row];
#pragma unroll
        for (int e = 0; e < 8; ++e) {
            int   d = seg * 8 + e;
            float v = (xs[row * DD + d] - m) * rs * nw[d] + nb[d];
            xbf[row * DD + d] = (__bf16)v;
        }
    }
    __syncthreads();

    // --- WMMA projections ---
    const int w  = t >> 5;        // wave id == h-tile
    const int l  = t & 31;
    const int lm = l & 15;
    const int kh = l >> 4;
    const int h  = w * 16 + lm;

    v8f acc0 = {}, acc1 = {}, acc2 = {}, acc3 = {}, acc4 = {};

#pragma unroll
    for (int kk = 0; kk < 4; ++kk) {
        v16bf a;
#pragma unroll
        for (int v = 0; v < 8; ++v) {
            int k0 = kk * 32 + kbase(v, kh);
            a[2 * v]     = xbf[lm * DD + k0];
            a[2 * v + 1] = xbf[lm * DD + k0 + 1];
        }
        v16bf b0, b1, b2, b3, b4;
#pragma unroll
        for (int v = 0; v < 8; ++v) {
            int k0 = kk * 32 + kbase(v, kh);
            b0[2 * v] = (__bf16)Wl [h * DD + k0]; b0[2 * v + 1] = (__bf16)Wl [h * DD + k0 + 1];
            b1[2 * v] = (__bf16)Wr [h * DD + k0]; b1[2 * v + 1] = (__bf16)Wr [h * DD + k0 + 1];
            b2[2 * v] = (__bf16)Wlg[h * DD + k0]; b2[2 * v + 1] = (__bf16)Wlg[h * DD + k0 + 1];
            b3[2 * v] = (__bf16)Wrg[h * DD + k0]; b3[2 * v + 1] = (__bf16)Wrg[h * DD + k0 + 1];
            b4[2 * v] = (__bf16)Wog[h * DD + k0]; b4[2 * v + 1] = (__bf16)Wog[h * DD + k0 + 1];
        }
        acc0 = wmma_bf16(a, b0, acc0);
        acc1 = wmma_bf16(a, b1, acc1);
        acc2 = wmma_bf16(a, b2, acc2);
        acc3 = wmma_bf16(a, b3, acc3);
        acc4 = wmma_bf16(a, b4, acc4);
    }

    // --- epilogue: C/D layout -> row = tile*16 + r + kh*8, col = h ---
#pragma unroll
    for (int r = 0; r < 8; ++r) {
        const long row = (long)tile * 16 + r + kh * 8;
        const float mk = mask[row];
        float lv = acc0[r] * mk * sigmoidf_(acc2[r]);
        float rv = acc1[r] * mk * sigmoidf_(acc3[r]);
        leftp [row * HH + h] = (__bf16)lv;
        rightp[row * HH + h] = (__bf16)rv;
        ogp   [row * HH + h] = sigmoidf_(acc4[r]);
    }
}

// ---------------------------------------------------------------------------
// K2: triangle einsum out[i,j,d] = sum_k L[i,k,d] * R[j,k,d]  (bf16 WMMA)
//   Block: 16x16 (i,j) tile, all 128 d channels, k-loop step 32.
//   256 KB dynamic LDS: Lt[16][32][128] + Rt[16][32][128] (bf16).
//   Wave w owns d in [16w, 16w+16) -> 16 f32 accumulator tiles.
// ---------------------------------------------------------------------------
__global__ __launch_bounds__(256) void trimul_einsum_kernel(
    const void* __restrict__ leftp_, const void* __restrict__ rightp_,
    float* __restrict__ op)
{
    const __bf16* __restrict__ leftp  = (const __bf16*)leftp_;
    const __bf16* __restrict__ rightp = (const __bf16*)rightp_;

    extern __shared__ char smem[];
    __bf16* Lt = (__bf16*)smem;             // [16][32][128]
    __bf16* Rt = Lt + 16 * 32 * DD;         // [16][32][128]

    const int t  = threadIdx.x;
    const int i0 = blockIdx.x * 16;
    const int j0 = blockIdx.y * 16;
    const int w  = t >> 5;
    const int l  = t & 31;
    const int lm = l & 15;
    const int kh = l >> 4;

    v8f acc[16] = {};

    for (int k0 = 0; k0 < NN; k0 += 32) {
        __syncthreads();
        // stage: 8192 uint4 per tile, 32 per thread, coalesced over d
#pragma unroll 8
        for (int jj = 0; jj < 32; ++jj) {
            int  li  = t + jj * 256;
            int  e   = li * 8;                   // bf16 element index
            int  i   = e >> 12;                  // / (32*128)
            int  rem = e & 4095;
            int  kk  = rem >> 7;
            int  d   = rem & 127;
            long gl  = ((long)(i0 + i) * NN + (k0 + kk)) * DD + d;
            long gr  = ((long)(j0 + i) * NN + (k0 + kk)) * DD + d;
            ((uint4*)Lt)[li] = *(const uint4*)(leftp  + gl);
            ((uint4*)Rt)[li] = *(const uint4*)(rightp + gr);
        }
        __syncthreads();

#pragma unroll
        for (int dd = 0; dd < 16; ++dd) {
            const int d = w * 16 + dd;
            v16bf a, b;
#pragma unroll
            for (int v = 0; v < 8; ++v) {
                int kk = kbase(v, kh);
                a[2 * v]     = Lt[lm * 4096 + kk * DD + d];
                a[2 * v + 1] = Lt[lm * 4096 + (kk + 1) * DD + d];
                b[2 * v]     = Rt[lm * 4096 + kk * DD + d];
                b[2 * v + 1] = Rt[lm * 4096 + (kk + 1) * DD + d];
            }
            acc[dd] = wmma_bf16(a, b, acc[dd]);
        }
    }

    // write o[i,j,d] (f32)
#pragma unroll
    for (int dd = 0; dd < 16; ++dd) {
        const int d = w * 16 + dd;
#pragma unroll
        for (int r = 0; r < 8; ++r) {
            int i = i0 + r + kh * 8;
            int j = j0 + lm;
            op[((long)i * NN + j) * DD + d] = acc[dd][r];
        }
    }
}

// ---------------------------------------------------------------------------
// K3: out = (LayerNorm(o) * out_gate) @ W_out.T
//   Same tiling as K1; single accumulator chain per wave.
// ---------------------------------------------------------------------------
__global__ __launch_bounds__(256) void trimul_out_kernel(
    const float* __restrict__ op, const float* __restrict__ ogp,
    const float* __restrict__ onw, const float* __restrict__ onb,
    const float* __restrict__ Wout, float* __restrict__ out)
{
    __shared__ float  xs[16 * HH];
    __shared__ float  part1[256], part2[256];
    __shared__ float  mu[16], rstd[16];
    __shared__ __bf16 vbf[16 * HH];

    const int  t    = threadIdx.x;
    const int  tile = blockIdx.x;
    const long base = (long)tile * 16 * HH;

    {
        const float4* g = (const float4*)(op + base);
        float4*       s = (float4*)xs;
        s[t]       = g[t];
        s[t + 256] = g[t + 256];
    }
    __syncthreads();
    {
        const int row = t >> 4, seg = t & 15;
        float s = 0.f, s2 = 0.f;
#pragma unroll
        for (int e = 0; e < 8; ++e) {
            float v = xs[row * HH + seg * 8 + e];
            s += v; s2 += v * v;
        }
        part1[row * 16 + seg] = s;
        part2[row * 16 + seg] = s2;
    }
    __syncthreads();
    if (t < 16) {
        float s = 0.f, s2 = 0.f;
#pragma unroll
        for (int e = 0; e < 16; ++e) { s += part1[t * 16 + e]; s2 += part2[t * 16 + e]; }
        float m   = s  * (1.0f / 128.0f);
        float var = s2 * (1.0f / 128.0f) - m * m;
        mu[t]   = m;
        rstd[t] = rsqrtf(var + 1e-5f);
    }
    __syncthreads();
    {
        const int row = t >> 4, seg = t & 15;
        const float m = mu[row], rs = rstd[row];
#pragma unroll
        for (int e = 0; e < 8; ++e) {
            int   d = seg * 8 + e;
            float v = (xs[row * HH + d] - m) * rs * onw[d] + onb[d];
            v *= ogp[base + row * HH + d];
            vbf[row * HH + d] = (__bf16)v;
        }
    }
    __syncthreads();

    const int w  = t >> 5;
    const int l  = t & 31;
    const int lm = l & 15;
    const int kh = l >> 4;
    const int dcol = w * 16 + lm;

    v8f acc = {};
#pragma unroll
    for (int kk = 0; kk < 4; ++kk) {
        v16bf a, b;
#pragma unroll
        for (int v = 0; v < 8; ++v) {
            int k0 = kk * 32 + kbase(v, kh);
            a[2 * v]     = vbf[lm * HH + k0];
            a[2 * v + 1] = vbf[lm * HH + k0 + 1];
            b[2 * v]     = (__bf16)Wout[dcol * HH + k0];
            b[2 * v + 1] = (__bf16)Wout[dcol * HH + k0 + 1];
        }
        acc = wmma_bf16(a, b, acc);
    }

#pragma unroll
    for (int r = 0; r < 8; ++r) {
        long row = (long)tile * 16 + r + kh * 8;
        out[row * DD + dcol] = acc[r];
    }
}

// ---------------------------------------------------------------------------
// Host launcher. Workspace layout (bytes):
//   [0, 37748736)              left  bf16 [147456,128]
//   [37748736, 75497472)       right bf16 [147456,128]
//   [75497472, 150994944)      out_gate f32 [147456,128]
//   [150994944, 226492416)     o (einsum result) f32 [147456,128]
// ---------------------------------------------------------------------------
extern "C" void kernel_launch(void* const* d_in, const int* in_sizes, int n_in,
                              void* d_out, int out_size, void* d_ws, size_t ws_size,
                              hipStream_t stream)
{
    const float* x    = (const float*)d_in[0];
    const float* mask = (const float*)d_in[1];
    const float* nw   = (const float*)d_in[2];
    const float* nb   = (const float*)d_in[3];
    const float* Wl   = (const float*)d_in[4];
    const float* Wr   = (const float*)d_in[5];
    const float* Wlg  = (const float*)d_in[6];
    const float* Wrg  = (const float*)d_in[7];
    const float* Wog  = (const float*)d_in[8];
    const float* onw  = (const float*)d_in[9];
    const float* onb  = (const float*)d_in[10];
    const float* Wout = (const float*)d_in[11];
    float*       out  = (float*)d_out;

    char* ws = (char*)d_ws;
    void*  leftp  = (void*)(ws);
    void*  rightp = (void*)(ws + (size_t)ROWS * HH * 2);
    float* ogp    = (float*)(ws + (size_t)ROWS * HH * 4);
    float* op     = (float*)(ws + (size_t)ROWS * HH * 8);

    trimul_proj_kernel<<<ROWS / 16, 256, 0, stream>>>(
        x, mask, nw, nb, Wl, Wr, Wlg, Wrg, Wog, leftp, rightp, ogp);

    // 256 KB dynamic LDS (<= 320 KB per WGP on CDNA5)
    trimul_einsum_kernel<<<dim3(NN / 16, NN / 16), 256, 262144, stream>>>(
        leftp, rightp, op);

    trimul_out_kernel<<<ROWS / 16, 256, 0, stream>>>(
        op, ogp, onw, onb, Wout, out);
}
